// MultiHeadedAttention_81466939671045
// MI455X (gfx1250) — matrix operand order repaired and checked by reading, not compiled, not run on hardware
//
#include <hip/hip_runtime.h>
#include <stdint.h>

typedef __attribute__((ext_vector_type(16))) __bf16 v16bf;
typedef __attribute__((ext_vector_type(8)))  float  v8f;
typedef unsigned int v4u __attribute__((ext_vector_type(4)));
typedef int          v8i __attribute__((ext_vector_type(8)));
typedef int          v4i __attribute__((ext_vector_type(4)));

#define SEQ   2048
#define EMB   1024
#define NHEAD 16
#define HD    64

// ---------------------------------------------------------------------------
// helpers
// ---------------------------------------------------------------------------
__device__ __forceinline__ void load8(const __bf16* p, v16bf& f, int base) {
  // 16-byte aligned load of 8 contiguous bf16 into fragment slots [base..base+7]
  union { uint4 u4; __bf16 b[8]; } cv;
  cv.u4 = *reinterpret_cast<const uint4*>(p);
#pragma unroll
  for (int e = 0; e < 8; ++e) f[base + e] = cv.b[e];
}

// ---------------------------------------------------------------------------
// Tensor Data Mover: DMA a 2D bf16 tile (tile_d0 x tile_d1 elements, row
// stride stride0 elements) from global memory into LDS at lds_addr, adding
// 32B of LDS padding after every 128B row (=> LDS row pitch 160B = stride-80
// bf16, matching the bank-spread layout the compute fragments expect).
//
// D# packing per CDNA5 ISA 8.3/8.4:
//   group0: [1:0]=count=1 | [63:32]=lds_addr | [120:64]=global_addr | type=2
//   group1: data_size=1(2B), pad_enable, pad_interval=4 (16 qwords=128B),
//           pad_amount=7 (8 dwords=32B), tensor_dim0/1, tile_dim0/1, stride0
// ---------------------------------------------------------------------------
__device__ __forceinline__ void tdm_load_2d(
    unsigned lds_addr, const void* gbase,
    unsigned tile_d0, unsigned tile_d1,
    unsigned tensor_d0, unsigned tensor_d1,
    unsigned stride0)
{
  const unsigned long long ga = (unsigned long long)(uintptr_t)gbase;
  v4u g0;
  g0[0] = 1u;                                       // count=1 (user descriptor)
  g0[1] = lds_addr;                                 // LDS byte address
  g0[2] = (unsigned)(ga & 0xFFFFFFFFu);             // global_addr[31:0]
  g0[3] = (unsigned)((ga >> 32) & 0x01FFFFFFu)      // global_addr[56:32]
        | (2u << 30);                               // type = 2 ("image")
  v8i g1;
  g1[0] = (int)((1u << 16)                          // data_size: 2 bytes
              | (1u << 20)                          // pad_enable
              | (4u << 22)                          // pad_interval: 16 qwords
              | (7u << 25));                        // pad_amount: 8 dwords
  g1[1] = (int)((tensor_d0 & 0xFFFFu) << 16);       // tensor_dim0[15:0]
  g1[2] = (int)((tensor_d0 >> 16) | ((tensor_d1 & 0xFFFFu) << 16));
  g1[3] = (int)((tensor_d1 >> 16) | (tile_d0 << 16));   // tile_dim0
  g1[4] = (int)(tile_d1 & 0xFFFFu);                 // tile_dim1 (tile_dim2=0)
  g1[5] = (int)stride0;                             // tensor_dim0_stride[31:0]
  g1[6] = 0;                                        // stride hi / dim1_stride lo
  g1[7] = 0;
  const v4i z4 = {0, 0, 0, 0};                      // groups 2/3 unused (2D)
  const v8i z8 = {0, 0, 0, 0, 0, 0, 0, 0};          // extra payload (unused)
  __builtin_amdgcn_tensor_load_to_lds(g0, g1, z4, z4, z8, 0);
}
__device__ __forceinline__ void wait_tensor_le2() {
  __builtin_amdgcn_s_wait_tensorcnt((short)2);
}
__device__ __forceinline__ void wait_tensor_le0() {
  __builtin_amdgcn_s_wait_tensorcnt((short)0);
}

// ---------------------------------------------------------------------------
// fp32 -> bf16 pre-conversion (memory bound)
// ---------------------------------------------------------------------------
__global__ __launch_bounds__(256) void cvt_f32_to_bf16(
    const float* __restrict__ src, __bf16* __restrict__ dst, int n)
{
  int i = (blockIdx.x * 256 + threadIdx.x) * 4;
  if (i < n) {
    float4 v = *reinterpret_cast<const float4*>(src + i);
    dst[i + 0] = (__bf16)v.x; dst[i + 1] = (__bf16)v.y;
    dst[i + 2] = (__bf16)v.z; dst[i + 3] = (__bf16)v.w;
  }
}

// ---------------------------------------------------------------------------
// GEMM:  C[M,N] = A[M,K] @ W[N,K]^T (+ bias), bf16 operands, f32 accumulate.
// 256 thr = 8 waves as 4(m) x 2(n); wave tile 32x64; block tile 128x128.
// K-step 64, double-buffered LDS tiles delivered by the Tensor Data Mover.
// ---------------------------------------------------------------------------
__global__ __launch_bounds__(256) void mha_gemm_wmma(
    const __bf16* __restrict__ A, const __bf16* __restrict__ Wb,
    const float* __restrict__ bias,
    float* __restrict__ C32, __bf16* __restrict__ C16,
    int M, int N, int K)
{
  __shared__ __align__(16) __bf16 sA[2][128][80];   // 40 KB (TDM pads to pitch 160B)
  __shared__ __align__(16) __bf16 sB[2][128][80];   // 40 KB

  const int tid  = threadIdx.x;
  const int w    = tid >> 5;
  const int lane = tid & 31;
  const int ln   = lane & 15;
  const int g    = lane >> 4;
  const int wm   = w & 3;          // 0..3
  const int wn   = w >> 2;         // 0..1
  const int bM   = blockIdx.x * 128;
  const int bN   = blockIdx.y * 128;

  v8f acc[2][4] = {};

  // wave 0 issues two TDM descriptors per k-tile (A 128x64, W 128x64)
  auto issue_tile = [&](int buf, int k0) {
    tdm_load_2d((unsigned)(uintptr_t)&sA[buf][0][0],
                A  + (size_t)bM * K + k0, 64, 128, 64, 128, (unsigned)K);
    tdm_load_2d((unsigned)(uintptr_t)&sB[buf][0][0],
                Wb + (size_t)bN * K + k0, 64, 128, 64, 128, (unsigned)K);
  };

  if (w == 0) issue_tile(0, 0);
  const int KT = K >> 6;
  for (int kt = 0; kt < KT; ++kt) {
    const int cur = kt & 1;
    if (w == 0) {
      if (kt + 1 < KT) {           // prefetch next tile into the other buffer
        issue_tile(cur ^ 1, (kt + 1) << 6);
        wait_tensor_le2();         // in-order: current tile's 2 DMAs retired
      } else {
        wait_tensor_le0();
      }
    }
    __syncthreads();

#pragma unroll
    for (int ks = 0; ks < 2; ++ks) {
      v16bf af[2], bfr[4];
#pragma unroll
      for (int i = 0; i < 2; ++i) {
        const __bf16* pa = &sA[cur][wm * 32 + i * 16 + ln][ks * 32];
        load8(pa + 8 * g,      af[i], 0);   // A layout: k = (e<8?e:e+8) + 8g
        load8(pa + 8 * g + 16, af[i], 8);
      }
#pragma unroll
      for (int j = 0; j < 4; ++j) {
        const __bf16* pb = &sB[cur][wn * 64 + j * 16 + ln][ks * 32];
        load8(pb + 8 * g,      bfr[j], 0);  // B layout: k = (e<8?e:e+8) + 8g
        load8(pb + 8 * g + 16, bfr[j], 8);
      }
#pragma unroll
      for (int i = 0; i < 2; ++i)
#pragma unroll
        for (int j = 0; j < 4; ++j)
          acc[i][j] = __builtin_amdgcn_wmma_f32_16x16x32_bf16(
              false, af[i], false, bfr[j], (short)0, acc[i][j], false, false);
    }
    __syncthreads();               // all reads of `cur` done before it refills
  }

  // store (C/D layout: row = r + 8g, col = ln within each 16x16 tile)
#pragma unroll
  for (int i = 0; i < 2; ++i) {
#pragma unroll
    for (int r = 0; r < 8; ++r) {
      int row = bM + wm * 32 + i * 16 + r + 8 * g;
#pragma unroll
      for (int j = 0; j < 4; ++j) {
        int col = bN + wn * 64 + j * 16 + ln;
        float v = acc[i][j][r];
        if (bias) v += bias[col];
        if (C32) C32[(size_t)row * N + col] = v;
        else     C16[(size_t)row * N + col] = (__bf16)v;
      }
    }
  }
}

// ---------------------------------------------------------------------------
// Flash attention over one (batch, head, 128-q-row block).
// Head slab = contiguous 128x1024 rows of the projection viewed as 2048x64.
// 8 waves x 16 q-rows; 64-key blocks; online softmax; P->LDS->A-layout->WMMA.
// K block arrives via TDM while V is register-staged + transposed.
// ---------------------------------------------------------------------------
__global__ __launch_bounds__(256) void mha_flash_attn(
    const __bf16* __restrict__ Qp, const __bf16* __restrict__ Kp,
    const __bf16* __restrict__ Vp, __bf16* __restrict__ Cc)
{
  __shared__ __align__(16) __bf16 sK [64][80];     // keys x dims        (10 KB)
  __shared__ __align__(16) __bf16 sVt[64][80];     // dims x keys        (10 KB)
  __shared__ __align__(16) __bf16 sP [8][16][80];  // per-wave P tile    (20 KB)

  const int tid  = threadIdx.x;
  const int w    = tid >> 5;
  const int lane = tid & 31;
  const int ln   = lane & 15;
  const int g    = lane >> 4;

  const int qblk = blockIdx.x;          // 0..15
  const int head = blockIdx.y;          // 0..15
  const int b    = blockIdx.z;

  const size_t headOff = ((size_t)b * SEQ + head * 128) * EMB;  // flat slab start
  const __bf16* Qh = Qp + headOff;      // view (2048, 64) row-major
  const __bf16* Kh = Kp + headOff;
  const __bf16* Vh = Vp + headOff;

  const int qbase = qblk * 128 + w * 16;

  // Q fragments (16 rows x 64 dims, two K=32 steps), resident in registers
  v16bf qf[2];
  {
    const __bf16* qrow = Qh + (size_t)(qbase + ln) * HD;
#pragma unroll
    for (int ks = 0; ks < 2; ++ks) {
      load8(qrow + 32 * ks + 8 * g,      qf[ks], 0);
      load8(qrow + 32 * ks + 8 * g + 16, qf[ks], 8);
    }
  }

  v8f o[4] = {};
  float run_max[8], run_sum[8];
#pragma unroll
  for (int r = 0; r < 8; ++r) { run_max[r] = -1e30f; run_sum[r] = 0.f; }

  const int nkb = 2 * qblk + 2;         // causal: keys up to qblk*128+127
  for (int kb = 0; kb < nkb; ++kb) {
    __syncthreads();
    // K block (64x64 bf16): single TDM descriptor issued by wave 0
    if (w == 0)
      tdm_load_2d((unsigned)(uintptr_t)&sK[0][0],
                  Kh + (size_t)kb * 64 * HD, 64, 64, 64, 64, HD);
    // V block: stage through registers (loads overlap the K DMA), then transpose
    union { uint4 u4; __bf16 bb[8]; } vreg[2];
#pragma unroll
    for (int i = 0; i < 2; ++i) {
      int idx = tid + i * 256;
      int row = idx >> 3;
      int c8  = (idx & 7) * 8;
      vreg[i].u4 = *reinterpret_cast<const uint4*>(Vh + (size_t)(kb * 64 + row) * HD + c8);
    }
#pragma unroll
    for (int i = 0; i < 2; ++i) {
      int idx = tid + i * 256;
      int row = idx >> 3;
      int c8  = (idx & 7) * 8;
#pragma unroll
      for (int t = 0; t < 8; ++t) sVt[c8 + t][row] = vreg[i].bb[t];
    }
    if (w == 0) wait_tensor_le0();
    __syncthreads();

    // S = (Q @ K^T) * 1/sqrt(64), with causal mask
    v8f sacc[4] = {};
#pragma unroll
    for (int ks = 0; ks < 2; ++ks) {
#pragma unroll
      for (int nt = 0; nt < 4; ++nt) {
        v16bf kf;
        const __bf16* pk = &sK[nt * 16 + ln][32 * ks + 8 * g];
        load8(pk,      kf, 0);
        load8(pk + 16, kf, 8);
        sacc[nt] = __builtin_amdgcn_wmma_f32_16x16x32_bf16(
            false, qf[ks], false, kf, (short)0, sacc[nt], false, false);
      }
    }
#pragma unroll
    for (int nt = 0; nt < 4; ++nt) {
      int j = kb * 64 + nt * 16 + ln;
#pragma unroll
      for (int r = 0; r < 8; ++r) {
        int q = qbase + r + 8 * g;
        float sv = sacc[nt][r] * 0.125f;
        sacc[nt][r] = (j > q) ? -1e30f : sv;
      }
    }

    // online softmax (rows spread over the 16 lanes of each half)
    float pcorr[8];
#pragma unroll
    for (int r = 0; r < 8; ++r) {
      float mx = fmaxf(fmaxf(sacc[0][r], sacc[1][r]), fmaxf(sacc[2][r], sacc[3][r]));
#pragma unroll
      for (int m = 8; m >= 1; m >>= 1) mx = fmaxf(mx, __shfl_xor(mx, m, 32));
      float nm   = fmaxf(run_max[r], mx);
      float corr = __expf(run_max[r] - nm);
      run_max[r] = nm;
      pcorr[r]   = corr;
      float rs = 0.f;
#pragma unroll
      for (int nt = 0; nt < 4; ++nt) {
        float p = __expf(sacc[nt][r] - nm);
        sacc[nt][r] = p;
        rs += p;
      }
#pragma unroll
      for (int m = 8; m >= 1; m >>= 1) rs += __shfl_xor(rs, m, 32);
      run_sum[r] = run_sum[r] * corr + rs;
    }
#pragma unroll
    for (int nt = 0; nt < 4; ++nt)
#pragma unroll
      for (int r = 0; r < 8; ++r)
        o[nt][r] *= pcorr[r];

    // P (D-layout) -> per-wave LDS tile (bf16)
#pragma unroll
    for (int nt = 0; nt < 4; ++nt)
#pragma unroll
      for (int r = 0; r < 8; ++r)
        sP[w][r + 8 * g][nt * 16 + ln] = (__bf16)sacc[nt][r];

    // O += P @ V   (reload P in A-layout, V from transposed tile in B-layout)
#pragma unroll
    for (int ks = 0; ks < 2; ++ks) {
      v16bf pf;
      const __bf16* pp = &sP[w][ln][32 * ks];
      load8(pp + 8 * g,      pf, 0);
      load8(pp + 8 * g + 16, pf, 8);
#pragma unroll
      for (int nt = 0; nt < 4; ++nt) {
        v16bf vf;
        const __bf16* pv = &sVt[nt * 16 + ln][32 * ks + 8 * g];
        load8(pv,      vf, 0);
        load8(pv + 16, vf, 8);
        o[nt] = __builtin_amdgcn_wmma_f32_16x16x32_bf16(
            false, pf, false, vf, (short)0, o[nt], false, false);
      }
    }
  }

  // finalize + scatter into concat[b, q, head*64 + d]
#pragma unroll
  for (int r = 0; r < 8; ++r) {
    float inv = 1.0f / run_sum[r];
    int q = qbase + r + 8 * g;
    size_t rowOff = ((size_t)b * SEQ + q) * EMB + head * HD;
#pragma unroll
    for (int nt = 0; nt < 4; ++nt)
      Cc[rowOff + nt * 16 + ln] = (__bf16)(o[nt][r] * inv);
  }
}

// ---------------------------------------------------------------------------
extern "C" void kernel_launch(void* const* d_in, const int* in_sizes, int n_in,
                              void* d_out, int out_size, void* d_ws, size_t ws_size,
                              hipStream_t stream) {
  (void)n_in; (void)out_size; (void)ws_size;
  const float* query = (const float*)d_in[0];
  const float* key   = (const float*)d_in[1];
  const float* value = (const float*)d_in[2];
  const float* W_Q   = (const float*)d_in[3];
  const float* W_K   = (const float*)d_in[4];
  const float* W_V   = (const float*)d_in[5];
  const float* W_ffn = (const float*)d_in[6];
  const float* b_ffn = (const float*)d_in[7];
  float* out = (float*)d_out;

  const int B  = in_sizes[0] / (SEQ * EMB);  // 2
  const int M  = B * SEQ;                    // 4096
  const int NX = M * EMB;                    // input elems
  const int NW = EMB * EMB;                  // weight elems

  // workspace layout (bf16): Xq,Xk,Xv | Wq,Wk,Wv,Wf | Qp,Kp,Vp | concat
  __bf16* Xq = (__bf16*)d_ws;
  __bf16* Xk = Xq + (size_t)NX;
  __bf16* Xv = Xk + (size_t)NX;
  __bf16* Wq = Xv + (size_t)NX;
  __bf16* Wk = Wq + (size_t)NW;
  __bf16* Wv = Wk + (size_t)NW;
  __bf16* Wf = Wv + (size_t)NW;
  __bf16* Qp = Wf + (size_t)NW;
  __bf16* Kp = Qp + (size_t)NX;
  __bf16* Vp = Kp + (size_t)NX;
  __bf16* Cc = Vp + (size_t)NX;

  dim3 blk(256);

  // 1) pre-convert every GEMM operand to bf16
  cvt_f32_to_bf16<<<dim3(NX / 1024), blk, 0, stream>>>(query, Xq, NX);
  cvt_f32_to_bf16<<<dim3(NX / 1024), blk, 0, stream>>>(key,   Xk, NX);
  cvt_f32_to_bf16<<<dim3(NX / 1024), blk, 0, stream>>>(value, Xv, NX);
  cvt_f32_to_bf16<<<dim3(NW / 1024), blk, 0, stream>>>(W_Q,   Wq, NW);
  cvt_f32_to_bf16<<<dim3(NW / 1024), blk, 0, stream>>>(W_K,   Wk, NW);
  cvt_f32_to_bf16<<<dim3(NW / 1024), blk, 0, stream>>>(W_V,   Wv, NW);
  cvt_f32_to_bf16<<<dim3(NW / 1024), blk, 0, stream>>>(W_ffn, Wf, NW);

  // 2) Q/K/V projections (bf16 -> bf16)
  dim3 ggrid(M / 128, EMB / 128);
  mha_gemm_wmma<<<ggrid, blk, 0, stream>>>(Xq, Wq, (const float*)nullptr,
                                           (float*)nullptr, Qp, M, EMB, EMB);
  mha_gemm_wmma<<<ggrid, blk, 0, stream>>>(Xk, Wk, (const float*)nullptr,
                                           (float*)nullptr, Kp, M, EMB, EMB);
  mha_gemm_wmma<<<ggrid, blk, 0, stream>>>(Xv, Wv, (const float*)nullptr,
                                           (float*)nullptr, Vp, M, EMB, EMB);

  // 3) attention: grid (qblock, head, batch)
  dim3 agrid(SEQ / 128, NHEAD, B);
  mha_flash_attn<<<agrid, blk, 0, stream>>>(Qp, Kp, Vp, Cc);

  // 4) final FFN (+bias, fp32 out)
  mha_gemm_wmma<<<ggrid, blk, 0, stream>>>(Cc, Wf, b_ffn,
                                           out, (__bf16*)nullptr, M, EMB, EMB);
}
